// PMA_62191126446175
// MI455X (gfx1250) — matrix operand order repaired and checked by reading, not compiled, not run on hardware
//
#include <hip/hip_runtime.h>
#include <hip/hip_bf16.h>
#include <math.h>

// ---------------------------------------------------------------------------
// PMA block for MI455X (gfx1250, wave32, WMMA + async-to-LDS pipeline).
//   Xh = f16(X)                                   (one-time cast, enables async)
//   Q  = S@Wq + bq                                 [32,2048]
//   Qk[hs,e] = sum_d Q[s,h*256+d] Wk[e,h*256+d]    [256,2048] f16
//   qb[hs]   = sum_c Q[s,c] bk[c]
//   scores[n,hs] = (Xh[n,:]·Qk[hs,:] + qb)/sqrt(D) <- WMMA GEMM, async dbl-buf
//   softmax over n per hs: Aht[hs,n] = exp (f16, transposed), colsum[hs]
//   rowX[hs,e] = (sum_n Aht[hs,n] Xh[n,e])/colsum  <- WMMA GEMM, async A-tiles
//   qq = Q + rowX@Wv_head + bv ; O = LN0(qq); O += relu(O@Wo+bo); out = LN1
// Workspace: ~93 MB of d_ws.
// ---------------------------------------------------------------------------

constexpr int kDim   = 2048;
constexpr int kHeads = 8;
constexpr int kSeeds = 32;
constexpr int kN     = 16384;
constexpr int kHS    = kHeads * kSeeds;   // 256
constexpr int kDS    = kDim / kHeads;     // 256
#define LN_EPS 1e-3f

typedef __attribute__((ext_vector_type(16))) _Float16 v16h;
typedef __attribute__((ext_vector_type(8)))  float    v8f;

union Frag16   { v16h v; float4 f4[2]; };
union HalfPack { float4 f4; _Float16 h[8]; };

// ---- CDNA5 async global->LDS copy (16B per enabled lane) -------------------
__device__ __forceinline__ void async_cp16(const void* g, void* lds) {
  asm volatile("global_load_async_to_lds_b128 %0, %1, off"
               :: "v"((unsigned)(uintptr_t)lds), "v"(g) : "memory");
}
__device__ __forceinline__ void wait_async0() {
  asm volatile("s_wait_asynccnt 0" ::: "memory");
}

// ---------------- one-time fp32 -> f16 cast of X ----------------------------

__global__ __launch_bounds__(256)
void k_xcast(const float* __restrict__ X, _Float16* __restrict__ Xh) {
  size_t u = (size_t)blockIdx.x * blockDim.x + threadIdx.x;   // 8 elems/thread
  const float4 a = *reinterpret_cast<const float4*>(X + u * 8);
  const float4 b = *reinterpret_cast<const float4*>(X + u * 8 + 4);
  HalfPack p;
  p.h[0] = (_Float16)a.x; p.h[1] = (_Float16)a.y;
  p.h[2] = (_Float16)a.z; p.h[3] = (_Float16)a.w;
  p.h[4] = (_Float16)b.x; p.h[5] = (_Float16)b.y;
  p.h[6] = (_Float16)b.z; p.h[7] = (_Float16)b.w;
  *reinterpret_cast<float4*>(Xh + u * 8) = p.f4;
}

// ---------------- small projection kernels (fp32 VALU) ----------------------

__global__ void k_qproj(const float* __restrict__ S, const float* __restrict__ Wq,
                        const float* __restrict__ bq, float* __restrict__ Q) {
  int idx = blockIdx.x * blockDim.x + threadIdx.x;   // 65536
  int s = idx >> 11, c = idx & 2047;
  float acc = bq[c];
  const float* srow = S + s * kDim;
  for (int e = 0; e < kDim; ++e) acc = fmaf(srow[e], Wq[e * kDim + c], acc);
  Q[idx] = acc;
}

__global__ void k_qb(const float* __restrict__ Q, const float* __restrict__ bk,
                     float* __restrict__ qb) {
  int hs = threadIdx.x;
  int s = hs & 31;
  float acc = 0.f;
  for (int c = 0; c < kDim; ++c) acc = fmaf(Q[s * kDim + c], bk[c], acc);
  qb[hs] = acc;
}

__global__ void k_qkprep(const float* __restrict__ Q, const float* __restrict__ Wk,
                         _Float16* __restrict__ Qkh) {
  int idx = blockIdx.x * blockDim.x + threadIdx.x;   // 524288
  int hs = idx >> 11, e = idx & 2047;
  int h = hs >> 5, s = hs & 31;
  const float* qrow = Q  + s * kDim + h * kDS;
  const float* wrow = Wk + e * kDim + h * kDS;
  float acc = 0.f;
  for (int d = 0; d < kDS; ++d) acc = fmaf(qrow[d], wrow[d], acc);
  Qkh[idx] = (_Float16)acc;
}

// -------------- GEMM 1: scores = Xh @ Qk^T -- async double-buffered ---------
// 256 thr = 8 waves; block tile 256(M=n) x 64(N=hs); wave w owns 32 M-rows
// (2 A-frags) x all 64 N (4 B-frags) -> 8 wmma per K-step, frags reused.

__global__ __launch_bounds__(256)
void k_scores(const _Float16* __restrict__ Xh, const _Float16* __restrict__ Qkh,
              const float* __restrict__ qb, float* __restrict__ scores) {
  __shared__ _Float16 lA[2][256][40];   // [buf][m][k], 80B rows (16B aligned)
  __shared__ _Float16 lB[2][64][40];    // [buf][n(hs)][k]
  const int t = threadIdx.x;
  const int lane = t & 31, wave = t >> 5;
  const int m0 = blockIdx.x * 256;
  const int n0 = blockIdx.y * 64;

  v8f acc[2][4] = {};

  auto fill = [&](int buf, int k) {
    #pragma unroll
    for (int i = 0; i < 4; ++i) {                      // A: 1024 x16B units
      int u = t + i * 256;
      int row = u >> 2, seg = u & 3;
      async_cp16(Xh + (size_t)(m0 + row) * kDim + k + seg * 8,
                 &lA[buf][row][seg * 8]);
    }
    int row = t >> 2, seg = t & 3;                     // B: 256 units
    async_cp16(Qkh + (size_t)(n0 + row) * kDim + k + seg * 8,
               &lB[buf][row][seg * 8]);
  };

  fill(0, 0);
  wait_async0();
  __syncthreads();

  int cur = 0;
  for (int k = 0; k < kDim; k += 32) {
    if (k + 32 < kDim) fill(cur ^ 1, k + 32);          // async prefetch

    Frag16 a0, a1;
    const int mr = wave * 32 + (lane & 15);
    const int kb = (lane >> 4) * 8;
    a0.f4[0] = *reinterpret_cast<const float4*>(&lA[cur][mr][kb]);
    a0.f4[1] = *reinterpret_cast<const float4*>(&lA[cur][mr][kb + 16]);
    a1.f4[0] = *reinterpret_cast<const float4*>(&lA[cur][mr + 16][kb]);
    a1.f4[1] = *reinterpret_cast<const float4*>(&lA[cur][mr + 16][kb + 16]);

    #pragma unroll
    for (int j = 0; j < 4; ++j) {
      Frag16 b;
      const int nc  = j * 16 + (lane & 15);
      const int kb2 = (lane >> 4) * 8;
      b.f4[0] = *reinterpret_cast<const float4*>(&lB[cur][nc][kb2]);
      b.f4[1] = *reinterpret_cast<const float4*>(&lB[cur][nc][kb2 + 16]);
      acc[0][j] = __builtin_amdgcn_wmma_f32_16x16x32_f16(
          false, a0.v, false, b.v, (short)0, acc[0][j], false, false);
      acc[1][j] = __builtin_amdgcn_wmma_f32_16x16x32_f16(
          false, a1.v, false, b.v, (short)0, acc[1][j], false, false);
    }

    wait_async0();       // next tile landed (per-wave)
    __syncthreads();     // ... for every wave; prior reads of nxt buffer done
    cur ^= 1;
  }

  const float scale = 0.022097086912079608f;   // 1/sqrt(2048)
  #pragma unroll
  for (int mi = 0; mi < 2; ++mi) {
    #pragma unroll
    for (int j = 0; j < 4; ++j) {
      const int gn  = n0 + j * 16 + (lane & 15);
      const float qbj = qb[gn];
      #pragma unroll
      for (int i = 0; i < 8; ++i) {
        const int gm = m0 + wave * 32 + mi * 16 + i + ((lane >> 4) << 3);
        scores[(size_t)gm * kHS + gn] = (acc[mi][j][i] + qbj) * scale;
      }
    }
  }
}

// -------------- softmax over n per hs column; emit f16 transposed A ---------

__global__ __launch_bounds__(256)
void k_softmax(const float* __restrict__ scores, _Float16* __restrict__ Aht,
               float* __restrict__ colsum) {
  __shared__ float red[256];
  const int hs = blockIdx.x, t = threadIdx.x;
  float m = -1e30f;
  for (int n = t; n < kN; n += 256) m = fmaxf(m, scores[(size_t)n * kHS + hs]);
  red[t] = m; __syncthreads();
  for (int s2 = 128; s2 > 0; s2 >>= 1) {
    if (t < s2) red[t] = fmaxf(red[t], red[t + s2]);
    __syncthreads();
  }
  m = red[0]; __syncthreads();

  float sum = 0.f;
  for (int n = t; n < kN; n += 256) {
    float e = __expf(scores[(size_t)n * kHS + hs] - m);
    Aht[(size_t)hs * kN + n] = (_Float16)e;   // coalesced transposed write
    sum += e;
  }
  red[t] = sum; __syncthreads();
  for (int s2 = 128; s2 > 0; s2 >>= 1) {
    if (t < s2) red[t] += red[t + s2];
    __syncthreads();
  }
  if (t == 0) colsum[hs] = red[0];
}

// -------------- GEMM 2: rowX = (A @ Xh) / colsum ----------------------------
// block tile 64(M=hs) x 128(N=e); waves 4x2, each 16Mx64N: 1 A-frag + 4
// B-frags -> 4 wmma per K-step. A-tiles async (Aht is [M][K] in memory);
// B-tiles (need [e][n]) via f16 global loads + register transpose into LDS.

__global__ __launch_bounds__(256)
void k_rowx(const _Float16* __restrict__ Aht, const _Float16* __restrict__ Xh,
            const float* __restrict__ colsum, float* __restrict__ rowX) {
  __shared__ _Float16 lA[2][64][40];    // [buf][hs][k]
  __shared__ _Float16 lB[2][128][40];   // [buf][e][k]
  const int t = threadIdx.x;
  const int lane = t & 31, w = t >> 5;
  const int m0 = blockIdx.x * 64;       // hs
  const int n0 = blockIdx.y * 128;      // e
  const int wm = (w & 3) * 16, wn = (w >> 2) * 64;

  v8f acc[4] = {};

  auto fillA = [&](int buf, int k) {            // 256 x 16B async units
    int row = t >> 2, seg = t & 3;
    async_cp16(Aht + (size_t)(m0 + row) * kN + k + seg * 8,
               &lA[buf][row][seg * 8]);
  };
  auto fillB = [&](int buf, int k) {            // 8 halves along e, transpose
    #pragma unroll
    for (int i = 0; i < 2; ++i) {
      int u = t + i * 256;
      int r = u >> 4, c8 = u & 15;              // r: k-local, c8: e-chunk
      HalfPack p;
      p.f4 = *reinterpret_cast<const float4*>(
          Xh + (size_t)(k + r) * kDim + n0 + c8 * 8);
      #pragma unroll
      for (int j = 0; j < 8; ++j) lB[buf][c8 * 8 + j][r] = p.h[j];
    }
  };

  fillB(0, 0);
  fillA(0, 0);
  wait_async0();
  __syncthreads();

  int cur = 0;
  for (int k = 0; k < kN; k += 32) {
    if (k + 32 < kN) { fillA(cur ^ 1, k + 32); fillB(cur ^ 1, k + 32); }

    Frag16 a;
    const int mr = wm + (lane & 15);
    const int kb = (lane >> 4) * 8;
    a.f4[0] = *reinterpret_cast<const float4*>(&lA[cur][mr][kb]);
    a.f4[1] = *reinterpret_cast<const float4*>(&lA[cur][mr][kb + 16]);

    const int kb2 = (lane >> 4) * 8;
    #pragma unroll
    for (int j = 0; j < 4; ++j) {
      Frag16 b;
      const int nc = wn + j * 16 + (lane & 15);
      b.f4[0] = *reinterpret_cast<const float4*>(&lB[cur][nc][kb2]);
      b.f4[1] = *reinterpret_cast<const float4*>(&lB[cur][nc][kb2 + 16]);
      acc[j] = __builtin_amdgcn_wmma_f32_16x16x32_f16(
          false, a.v, false, b.v, (short)0, acc[j], false, false);
    }

    wait_async0();
    __syncthreads();
    cur ^= 1;
  }

  #pragma unroll
  for (int i = 0; i < 8; ++i) {
    const int gm = m0 + wm + i + ((lane >> 4) << 3);
    const float inv = 1.0f / colsum[gm];
    #pragma unroll
    for (int j = 0; j < 4; ++j) {
      const int gn = n0 + wn + j * 16 + (lane & 15);
      rowX[(size_t)gm * kDim + gn] = acc[j][i] * inv;
    }
  }
}

// -------------- per-head V projection on pooled rows + residual -------------

__global__ void k_qq(const float* __restrict__ Q, const float* __restrict__ rowX,
                     const float* __restrict__ Wv, const float* __restrict__ bv,
                     float* __restrict__ qq) {
  int idx = blockIdx.x * blockDim.x + threadIdx.x;   // 65536
  int s = idx >> 11, c = idx & 2047;
  int hs = ((c >> 8) << 5) + s;                      // h*32 + s
  const float* rx = rowX + (size_t)hs * kDim;
  float acc = bv[c];
  for (int e = 0; e < kDim; ++e) acc = fmaf(rx[e], Wv[e * kDim + c], acc);
  qq[idx] = Q[idx] + acc;
}

// -------------- LayerNorm ---------------------------------------------------

__global__ __launch_bounds__(256)
void k_ln(const float* __restrict__ x, const float* __restrict__ g,
          const float* __restrict__ b, float* __restrict__ y) {
  __shared__ float rs[256], rq[256];
  const int row = blockIdx.x, t = threadIdx.x;
  float s = 0.f, q = 0.f;
  for (int c = t; c < kDim; c += 256) {
    float v = x[(size_t)row * kDim + c];
    s += v; q += v * v;
  }
  rs[t] = s; rq[t] = q; __syncthreads();
  for (int k = 128; k > 0; k >>= 1) {
    if (t < k) { rs[t] += rs[t + k]; rq[t] += rq[t + k]; }
    __syncthreads();
  }
  const float mean = rs[0] * (1.0f / kDim);
  const float var  = rq[0] * (1.0f / kDim) - mean * mean;
  const float inv  = rsqrtf(var + LN_EPS);
  for (int c = t; c < kDim; c += 256)
    y[(size_t)row * kDim + c] = (x[(size_t)row * kDim + c] - mean) * inv * g[c] + b[c];
}

// -------------- MLP residual ------------------------------------------------

__global__ void k_mlp(const float* __restrict__ O, const float* __restrict__ Wo,
                      const float* __restrict__ bo, float* __restrict__ y) {
  int idx = blockIdx.x * blockDim.x + threadIdx.x;   // 65536
  int s = idx >> 11, c = idx & 2047;
  const float* orow = O + (size_t)s * kDim;
  float acc = bo[c];
  for (int e = 0; e < kDim; ++e) acc = fmaf(orow[e], Wo[e * kDim + c], acc);
  y[idx] = O[idx] + fmaxf(acc, 0.f);
}

// ---------------------------------------------------------------------------

extern "C" void kernel_launch(void* const* d_in, const int* in_sizes, int n_in,
                              void* d_out, int out_size, void* d_ws, size_t ws_size,
                              hipStream_t stream) {
  (void)in_sizes; (void)n_in; (void)out_size; (void)ws_size;
  const float* X  = (const float*)d_in[0];
  const float* S  = (const float*)d_in[1];
  const float* Wq = (const float*)d_in[2];
  const float* bq = (const float*)d_in[3];
  const float* Wk = (const float*)d_in[4];
  const float* bk = (const float*)d_in[5];
  const float* Wv = (const float*)d_in[6];
  const float* bv = (const float*)d_in[7];
  const float* Wo = (const float*)d_in[8];
  const float* bo = (const float*)d_in[9];
  const float* g0 = (const float*)d_in[10];
  const float* b0 = (const float*)d_in[11];
  const float* g1 = (const float*)d_in[12];
  const float* b1 = (const float*)d_in[13];

  char* ws = (char*)d_ws;
  size_t off = 0;
  auto carve = [&](size_t bytes) {
    void* p = ws + off;
    off = (off + bytes + 255) & ~(size_t)255;
    return p;
  };
  _Float16* Xh     = (_Float16*)carve((size_t)kN * kDim * 2);       // 64 MB
  float*    Q      = (float*)   carve((size_t)kSeeds * kDim * 4);
  float*    qb     = (float*)   carve(kHS * 4);
  _Float16* Qkh    = (_Float16*)carve((size_t)kHS * kDim * 2);      // 1 MB
  float*    scores = (float*)   carve((size_t)kN * kHS * 4);        // 16 MB
  _Float16* Aht    = (_Float16*)carve((size_t)kHS * kN * 2);        // 8 MB
  float*    colsum = (float*)   carve(kHS * 4);
  float*    rowX   = (float*)   carve((size_t)kHS * kDim * 4);      // 2 MB
  float*    qq     = (float*)   carve((size_t)kSeeds * kDim * 4);
  float*    O0     = (float*)   carve((size_t)kSeeds * kDim * 4);
  float*    Op     = (float*)   carve((size_t)kSeeds * kDim * 4);

  k_xcast  <<<(kN * kDim / 8) / 256, 256, 0, stream>>>(X, Xh);
  k_qproj  <<<256, 256, 0, stream>>>(S, Wq, bq, Q);
  k_qb     <<<1,   256, 0, stream>>>(Q, bk, qb);
  k_qkprep <<<2048,256, 0, stream>>>(Q, Wk, Qkh);
  k_scores <<<dim3(kN / 256, kHS / 64), 256, 0, stream>>>(Xh, Qkh, qb, scores);
  k_softmax<<<kHS, 256, 0, stream>>>(scores, Aht, colsum);
  k_rowx   <<<dim3(kHS / 64, kDim / 128), 256, 0, stream>>>(Aht, Xh, colsum, rowX);
  k_qq     <<<256, 256, 0, stream>>>(Q, rowX, Wv, bv, qq);
  k_ln     <<<kSeeds, 256, 0, stream>>>(qq, g0, b0, O0);
  k_mlp    <<<256, 256, 0, stream>>>(O0, Wo, bo, Op);
  k_ln     <<<kSeeds, 256, 0, stream>>>(Op, g1, b1, (float*)d_out);
}
